// LSTMLayer_3899830304993
// MI455X (gfx1250) — matrix operand order repaired and checked by reading, not compiled, not run on hardware
//
#include <hip/hip_runtime.h>
#include <hip/hip_bf16.h>

// ---------------------------------------------------------------------------
// LSTM layer, bf16 WMMA (v_wmma_f32_16x16x32_bf16), fp32 accumulate.
// T=512, B=64, I=H=1024.
// Recurrence: per-step kernel, W_hh j-slice staged to LDS via async
// global->LDS copies, h carried between steps as bf16 (double-buffered).
// ---------------------------------------------------------------------------

typedef __bf16 v16bf __attribute__((ext_vector_type(16)));
typedef __bf16 v8bf  __attribute__((ext_vector_type(8)));
typedef __bf16 v4bf  __attribute__((ext_vector_type(4)));
typedef float  v8f   __attribute__((ext_vector_type(8)));
typedef float  v4f   __attribute__((ext_vector_type(4)));
typedef int    v4i   __attribute__((ext_vector_type(4)));

#define T_STEPS 512
#define BATCH   64
#define IN_DIM  1024
#define HID     1024
#define G4      (4 * HID)          // 4096 gate rows

__device__ __forceinline__ float sigmoid_fast(float x) {
    return 1.0f / (1.0f + __expf(-x));
}
__device__ __forceinline__ float tanh_fast(float x) {
    return 2.0f / (1.0f + __expf(-2.0f * x)) - 1.0f;
}

// ---- async global -> LDS 16-byte copy (gfx1250 ASYNCcnt path) -------------
// Builtin signature (from hipcc diagnostic): (global int4*, lds int4*, Ii, Ii)
__device__ __forceinline__ void copy16_g2lds(const __bf16* gp, __bf16* lp) {
#if __has_builtin(__builtin_amdgcn_global_load_async_to_lds_b128)
    __builtin_amdgcn_global_load_async_to_lds_b128(
        (__attribute__((address_space(1))) v4i*)gp,
        (__attribute__((address_space(3))) v4i*)lp, 0, 0);
#else
    *(v8bf*)lp = *(const v8bf*)gp;   // fallback: global load + ds_store
#endif
}
__device__ __forceinline__ void wait_async0() {
#if __has_builtin(__builtin_amdgcn_s_wait_asynccnt)
    __builtin_amdgcn_s_wait_asynccnt(0);
#else
    asm volatile("s_wait_asynccnt 0x0" ::: "memory");
#endif
}

// A fragment from bf16 row-major source: 16x32 tile.
// lanes 0-15: row m0+lane, K = kbase + 0..15
// lanes 16-31: row m0+lane-16, K = kbase + 16..31
__device__ __forceinline__ v16bf load_a_frag_bf16(const __bf16* __restrict__ A,
                                                  int m0, int kbase, int lane) {
    int row   = m0 + (lane & 15);
    int koff  = kbase + ((lane >> 4) & 1) * 16;
    const __bf16* p = A + (size_t)row * HID + koff;
    v8bf lo = *(const v8bf*)(p + 0);    // K +0..7
    v8bf hi = *(const v8bf*)(p + 8);    // K +8..15
    v16bf a;
#pragma unroll
    for (int i = 0; i < 8; ++i) { a[i] = lo[i]; a[8 + i] = hi[i]; }
    return a;
}

// A fragment with on-the-fly fp32 -> bf16 conversion (xproj path).
__device__ __forceinline__ v16bf load_a_frag_f32(const float* __restrict__ A,
                                                 int lda, int m0, int kbase,
                                                 int lane) {
    int row   = m0 + (lane & 15);
    int khalf = (lane >> 4) & 1;
    const float* p = A + (size_t)row * lda + kbase + khalf * 16;
    v4f f0 = *(const v4f*)(p + 0);
    v4f f1 = *(const v4f*)(p + 4);
    v4f f2 = *(const v4f*)(p + 8);
    v4f f3 = *(const v4f*)(p + 12);
    v16bf a;
#pragma unroll
    for (int i = 0; i < 4; ++i) {
        a[i]      = (__bf16)f0[i];
        a[4 + i]  = (__bf16)f1[i];
        a[8 + i]  = (__bf16)f2[i];
        a[12 + i] = (__bf16)f3[i];
    }
    return a;
}

// B fragment: 32x16 tile, B(k,n) = W[n,k]; W bf16 row-major, row stride 1024.
// lanes 0-15: col n0+lane, K = kbase+0..15 ; lanes 16-31: K = kbase+16..31.
__device__ __forceinline__ v16bf load_b_frag_bf16(const __bf16* __restrict__ W,
                                                  int n0, int kbase, int lane) {
    int nrow = n0 + (lane & 15);
    int koff = kbase + ((lane >> 4) & 1) * 16;
    const __bf16* p = W + (size_t)nrow * IN_DIM + koff;
    v8bf lo = *(const v8bf*)(p + 0);
    v8bf hi = *(const v8bf*)(p + 8);
    v16bf b;
#pragma unroll
    for (int i = 0; i < 8; ++i) { b[i] = lo[i]; b[8 + i] = hi[i]; }
    return b;
}

// --------------------------- weight conversion -----------------------------
__global__ __launch_bounds__(256) void cvt_weights(
    const float* __restrict__ Wih, const float* __restrict__ Whh,
    __bf16* __restrict__ out) {
    const size_t HALF = (size_t)G4 * IN_DIM;  // 4194304
    size_t idx = ((size_t)blockIdx.x * blockDim.x + threadIdx.x) * 4;
    const float* src = (idx < HALF) ? (Wih + idx) : (Whh + (idx - HALF));
    v4f f = *(const v4f*)src;
    v4bf o;
#pragma unroll
    for (int i = 0; i < 4; ++i) o[i] = (__bf16)f[i];
    *(v4bf*)(out + idx) = o;
}

// ----------------------------- state init ----------------------------------
__global__ __launch_bounds__(256) void init_state(const float* __restrict__ c0,
                                                  const float* __restrict__ h0,
                                                  float* __restrict__ cstate,
                                                  __bf16* __restrict__ hbf0) {
    size_t idx = ((size_t)blockIdx.x * blockDim.x + threadIdx.x) * 4;
    *(v4f*)(cstate + idx) = *(const v4f*)(c0 + idx);
    v4f h = *(const v4f*)(h0 + idx);
    v4bf hb;
#pragma unroll
    for (int i = 0; i < 4; ++i) hb[i] = (__bf16)h[i];
    *(v4bf*)(hbf0 + idx) = hb;
}

// --------------------------- input projection ------------------------------
// XP[32768,4096] = X[32768,1024] @ Wih^T. One wave = 16(M) x 128(N) strip.
__global__ __launch_bounds__(256) void xproj_gemm(
    const float* __restrict__ X, const __bf16* __restrict__ Wih,
    float* __restrict__ XP) {
    int lane = threadIdx.x & 31;
    int wid  = blockIdx.x * (blockDim.x >> 5) + (threadIdx.x >> 5);
    int mtile  = wid >> 5;
    int nstrip = wid & 31;
    int m0 = mtile * 16;
    int n0 = nstrip * 128;

    v8f acc[8];
#pragma unroll
    for (int i = 0; i < 8; ++i) acc[i] = (v8f)0.0f;

    for (int kt = 0; kt < IN_DIM / 32; ++kt) {
        v16bf a = load_a_frag_f32(X, IN_DIM, m0, kt * 32, lane);
#pragma unroll
        for (int nt = 0; nt < 8; ++nt) {
            v16bf b = load_b_frag_bf16(Wih, n0 + nt * 16, kt * 32, lane);
            acc[nt] = __builtin_amdgcn_wmma_f32_16x16x32_bf16(
                false, a, false, b, (short)0, acc[nt], false, false);
        }
    }

    int N  = lane & 15;
    int mo = (lane >> 4) * 8;
#pragma unroll
    for (int nt = 0; nt < 8; ++nt) {
#pragma unroll
        for (int r = 0; r < 8; ++r) {
            int row = m0 + mo + r;
            XP[(size_t)row * G4 + n0 + nt * 16 + N] = acc[nt][r];
        }
    }
}

// ------------------------------ LSTM step ----------------------------------
// One 128-thread block owns one j-tile. Its 4 waves cover the 4 batch tiles.
// The 64 W_hh rows this block needs (4 gates x 16 rows, each 1024 k) are
// async-copied to LDS once, then all 32 k-iterations read B frags from LDS.
__global__ __launch_bounds__(128) void lstm_step(
    const __bf16* __restrict__ hprev_bf,  // [64,1024] bf16
    const __bf16* __restrict__ Whh,       // [4096,1024] bf16
    const float* __restrict__ XPt,        // x_proj slice for t: [64,4096]
    const float* __restrict__ b_ih, const float* __restrict__ b_hh,
    float* __restrict__ cstate,           // [64,1024] fp32, in place
    float* __restrict__ hout,             // [64,1024] fp32 -> outputs[t]
    __bf16* __restrict__ hnext_bf) {      // [64,1024] bf16 for step t+1
    // local row r = g*16 + jr  ->  Whh row g*1024 + j0 + jr
    __shared__ __attribute__((aligned(16))) __bf16 sW[64 * 1024];  // 128 KB

    int tid  = threadIdx.x;
    int lane = tid & 31;
    int wv   = tid >> 5;            // 0..3 = batch tile
    int j0   = blockIdx.x * 16;
    int m0   = wv * 16;

    // ---- stage W_hh slice: 8192 16-byte chunks, 64 per thread, all async --
    for (int c = tid; c < 8192; c += 128) {
        int r  = c >> 7;            // local row (128 chunks of 16B per 2KB row)
        int ck = c & 127;
        int grow = (r >> 4) * HID + j0 + (r & 15);
        copy16_g2lds(Whh + (size_t)grow * IN_DIM + ck * 8, &sW[r * 1024 + ck * 8]);
    }
    wait_async0();
    __syncthreads();

    v8f acc[4];
#pragma unroll
    for (int g = 0; g < 4; ++g) acc[g] = (v8f)0.0f;

    for (int kt = 0; kt < HID / 32; ++kt) {
        v16bf a = load_a_frag_bf16(hprev_bf, m0, kt * 32, lane);
        int koff = kt * 32 + ((lane >> 4) & 1) * 16;
#pragma unroll
        for (int g = 0; g < 4; ++g) {
            const __bf16* p = &sW[(g * 16 + (lane & 15)) * 1024 + koff];
            v8bf lo = *(const v8bf*)(p + 0);
            v8bf hi = *(const v8bf*)(p + 8);
            v16bf b;
#pragma unroll
            for (int i = 0; i < 8; ++i) { b[i] = lo[i]; b[8 + i] = hi[i]; }
            acc[g] = __builtin_amdgcn_wmma_f32_16x16x32_bf16(
                false, a, false, b, (short)0, acc[g], false, false);
        }
    }

    int N  = lane & 15;
    int mo = (lane >> 4) * 8;
    int n  = j0 + N;

    float bias[4];
#pragma unroll
    for (int g = 0; g < 4; ++g) bias[g] = b_ih[g * HID + n] + b_hh[g * HID + n];

#pragma unroll
    for (int r = 0; r < 8; ++r) {
        int row = m0 + mo + r;
        size_t gbase = (size_t)row * G4;
        float gi = acc[0][r] + XPt[gbase + 0 * HID + n] + bias[0];
        float gf = acc[1][r] + XPt[gbase + 1 * HID + n] + bias[1];
        float gg = acc[2][r] + XPt[gbase + 2 * HID + n] + bias[2];
        float go = acc[3][r] + XPt[gbase + 3 * HID + n] + bias[3];
        size_t cidx = (size_t)row * HID + n;
        float c_new = sigmoid_fast(gf) * cstate[cidx] +
                      sigmoid_fast(gi) * tanh_fast(gg);
        cstate[cidx] = c_new;
        float h_new  = sigmoid_fast(go) * tanh_fast(c_new);
        hout[cidx]     = h_new;
        hnext_bf[cidx] = (__bf16)h_new;
    }
}

// ------------------------------- tail --------------------------------------
__global__ __launch_bounds__(256) void tail_copy(float* __restrict__ out,
                                                 const float* __restrict__ cstate) {
    const size_t TBH = (size_t)T_STEPS * BATCH * HID;
    const size_t BH  = (size_t)BATCH * HID;
    size_t i = (size_t)blockIdx.x * blockDim.x + threadIdx.x;
    out[TBH + i]      = out[(T_STEPS - 1) * BH + i];  // final h
    out[TBH + BH + i] = cstate[i];                    // final c
}

// ---------------------------------------------------------------------------
extern "C" void kernel_launch(void* const* d_in, const int* in_sizes, int n_in,
                              void* d_out, int out_size, void* d_ws, size_t ws_size,
                              hipStream_t stream) {
    const float* inp  = (const float*)d_in[0];  // [512,64,1024]
    const float* h0   = (const float*)d_in[1];  // [64,1024]
    const float* c0   = (const float*)d_in[2];  // [64,1024]
    const float* W_ih = (const float*)d_in[3];  // [4096,1024]
    const float* W_hh = (const float*)d_in[4];  // [4096,1024]
    const float* b_ih = (const float*)d_in[5];  // [4096]
    const float* b_hh = (const float*)d_in[6];  // [4096]
    float* out = (float*)d_out;

    char* ws = (char*)d_ws;
    __bf16* Wih_b  = (__bf16*)ws;                                  //  8 MB
    __bf16* Whh_b  = (__bf16*)(ws + ((size_t)8 << 20));            //  8 MB
    float*  cstate = (float*)(ws + ((size_t)16 << 20));            // 256 KB
    __bf16* hbf0   = (__bf16*)(ws + ((size_t)16 << 20) + (256u << 10)); // 128 KB
    __bf16* hbf1   = (__bf16*)(ws + ((size_t)16 << 20) + (384u << 10)); // 128 KB
    float*  xproj  = (float*)(ws + ((size_t)17 << 20));            // 512 MB

    const size_t BH = (size_t)BATCH * HID;

    // 1) weights fp32 -> bf16
    cvt_weights<<<8192, 256, 0, stream>>>(W_ih, W_hh, Wih_b);

    // 2) c state init + h0 -> bf16
    init_state<<<64, 256, 0, stream>>>(c0, h0, cstate, hbf0);

    // 3) big parallel input-projection GEMM
    xproj_gemm<<<8192, 256, 0, stream>>>(inp, Wih_b, xproj);

    // 4) sequential recurrence; h flows as bf16 (double-buffered in ws),
    //    fp32 h rows land directly in d_out.
    for (int t = 0; t < T_STEPS; ++t) {
        const __bf16* hp = (t & 1) ? hbf1 : hbf0;
        __bf16*       hn = (t & 1) ? hbf0 : hbf1;
        lstm_step<<<64, 128, 0, stream>>>(
            hp, Whh_b, xproj + (size_t)t * BATCH * G4,
            b_ih, b_hh, cstate, out + (size_t)t * BH, hn);
    }

    // 5) final h, c to output tail
    tail_copy<<<256, 256, 0, stream>>>(out, cstate);
}
